// GRUModel_8735963480655
// MI455X (gfx1250) — matrix operand order repaired and checked by reading, not compiled, not run on hardware
//
#include <hip/hip_runtime.h>

// ---------------------------------------------------------------------------
// GRU model for MI455X (gfx1250, wave32, WMMA).
//   V=50000, E=300 (padded to 320), H=1024, O=5, B=64, T=512
// Pipeline (all on `stream`):
//   k_init      : zero h0 (f32 + bf16) and grid-barrier words in ws
//   k_pack_wih  : W_ih [3072,300] f32 -> [3072,320] bf16 (zero-padded K)
//   k_pack_whh  : W_hh [3072,1024] f32 -> bf16
//   k_embed     : gather + bf16 pack  -> A_emb [32768,320] bf16
//   k_gemm_gx   : gx = A_emb @ W_ih^T + b_ih  (f32 [32768,3072]) via WMMA bf16
//   k_gru       : persistent cooperative kernel, 512 sequential steps:
//                   gh = h @ W_hh^T + b_hh  (768 waves x 32 wmma)  | grid barrier
//                   gates -> h (f32 + bf16)                        | grid barrier
//   k_fc        : out = h @ W_fc^T + b_fc  (64x5)
// ---------------------------------------------------------------------------

typedef __attribute__((ext_vector_type(16))) __bf16 v16bf;
typedef __attribute__((ext_vector_type(8)))  __bf16 v8bf;
typedef __attribute__((ext_vector_type(8)))  float  v8f;

#define Bsz   64
#define Tlen  512
#define Edim  300
#define Epad  320
#define Hdim  1024
#define G3H   3072
#define Odim  5
#define NROW  (Bsz * Tlen)        // 32768

#define NUM_WG_GRU 96             // 96 WG * 8 waves = 768 waves = 4 x 192 tiles

// ---------------- workspace layout (bytes, all 256-aligned) ----------------
#define OFF_BAR   ((size_t)0)                                    // 2 uints
#define OFF_HF    ((size_t)256)                                  // h f32  64x1024
#define OFF_HB    (OFF_HF  + (size_t)Bsz * Hdim * 4)             // h bf16 64x1024
#define OFF_GH    (OFF_HB  + (size_t)Bsz * Hdim * 2)             // gh f32 64x3072
#define OFF_WIH   (OFF_GH  + (size_t)Bsz * G3H * 4)              // W_ih bf16 3072x320
#define OFF_WHH   (OFF_WIH + (size_t)G3H * Epad * 2)             // W_hh bf16 3072x1024
#define OFF_AEMB  (OFF_WHH + (size_t)G3H * Hdim * 2)             // A_emb bf16 32768x320
#define OFF_GX    (OFF_AEMB + (size_t)NROW * Epad * 2)           // gx f32 32768x3072

// ---------------- WMMA fragment helpers (ISA 7.12.2 layouts) ----------------
// A: 16x32 bf16, row-major source, lda elements per row.
//   lanes 0-15: row m0+lane,   K {k0..k0+7} in v0-3, {k0+16..k0+23} in v4-7
//   lanes16-31: row m0+lane-16,K {k0+8..15} in v0-3, {k0+24..31}    in v4-7
__device__ __forceinline__ v16bf load_a_frag(const __bf16* A, int lda,
                                             int m0, int k0, int lane) {
  int r  = m0 + (lane & 15);
  int kh = (lane >> 4) << 3;                       // 0 or 8
  const __bf16* p = A + (size_t)r * lda + k0 + kh;
  v8bf lo = *(const v8bf*)(p);                     // 16B
  v8bf hi = *(const v8bf*)(p + 16);                // 16B
  v16bf a;
#pragma unroll
  for (int i = 0; i < 8; ++i) { a[i] = lo[i]; a[i + 8] = hi[i]; }
  return a;
}

// B: 32x16 bf16. We hold weights as W[N,K] row-major (= B column-major),
// so column n of B is row n of W. lanes 0-15: col n0+lane, K {k0..k0+7},
// {k0+16..k0+23}; lanes 16-31: col n0+lane-16, K {k0+8..15}, {k0+24..31}.
__device__ __forceinline__ v16bf load_b_frag(const __bf16* W, int ldw,
                                             int n0, int k0, int lane) {
  int n  = n0 + (lane & 15);
  int kb = k0 + ((lane >> 4) << 3);                // 0 or 8
  const __bf16* p = W + (size_t)n * ldw + kb;
  v8bf lo = *(const v8bf*)(p);                     // 16B
  v8bf hi = *(const v8bf*)(p + 16);                // 16B
  v16bf b;
#pragma unroll
  for (int i = 0; i < 8; ++i) { b[i] = lo[i]; b[i + 8] = hi[i]; }
  return b;
}

// C/D: v8f; VGPR r: lanes 0-15 -> (M=r, N=lane), lanes 16-31 -> (M=r+8, N-16)
__device__ __forceinline__ void store_c_bias(float* D, int ldd, int m0, int n0,
                                             int lane, v8f c, float bias) {
  int col = n0 + (lane & 15);
  int rb  = m0 + ((lane >> 4) << 3);
#pragma unroll
  for (int r = 0; r < 8; ++r)
    D[(size_t)(rb + r) * ldd + col] = c[r] + bias;
}

__device__ __forceinline__ float sigmoidf_(float x) {
  return 1.0f / (1.0f + __expf(-x));
}

// ------------------------------- init ---------------------------------------
__global__ void k_init(float* hf, __bf16* hb, unsigned* bar) {
  if (blockIdx.x == 0 && threadIdx.x < 2) bar[threadIdx.x] = 0u;
  for (int i = blockIdx.x * blockDim.x + threadIdx.x; i < Bsz * Hdim;
       i += gridDim.x * blockDim.x) {
    hf[i] = 0.0f;
    hb[i] = (__bf16)0.0f;
  }
}

// --------------------------- weight packing ---------------------------------
__global__ void k_pack_wih(const float* W, __bf16* Wp) {   // [3072,300]->[3072,320]
  for (int i = blockIdx.x * blockDim.x + threadIdx.x; i < G3H * Epad;
       i += gridDim.x * blockDim.x) {
    int g = i / Epad, k = i % Epad;
    Wp[i] = (__bf16)((k < Edim) ? W[(size_t)g * Edim + k] : 0.0f);
  }
}

__global__ void k_pack_whh(const float* W, __bf16* Wp) {   // [3072,1024]
  for (int i = blockIdx.x * blockDim.x + threadIdx.x; i < G3H * Hdim;
       i += gridDim.x * blockDim.x)
    Wp[i] = (__bf16)W[i];
}

// --------------------- embedding gather + bf16 pack --------------------------
__global__ void k_embed(const int* __restrict__ x, const float* __restrict__ emb,
                        __bf16* __restrict__ Ae) {
  for (size_t i = (size_t)blockIdx.x * blockDim.x + threadIdx.x;
       i < (size_t)NROW * Epad; i += (size_t)gridDim.x * blockDim.x) {
    int row = (int)(i / Epad);
    int e   = (int)(i % Epad);
    int xi  = x[row];                                  // row = b*T + t
    float v = (e < Edim) ? emb[(size_t)xi * Edim + e] : 0.0f;
    Ae[i] = (__bf16)v;
  }
}

// --------------- phase 1: gx = A_emb @ W_ih^T + b_ih  (WMMA) ----------------
// M=32768 (2048 tiles), N=3072 (192 tiles), K=320 (10 wmma steps)
__global__ void __launch_bounds__(256) k_gemm_gx(const __bf16* __restrict__ Ae,
                                                 const __bf16* __restrict__ Wih,
                                                 const float* __restrict__ b_ih,
                                                 float* __restrict__ gx) {
  const int lane = threadIdx.x & 31;
  const int wave = threadIdx.x >> 5;
  const long tile = (long)blockIdx.x * 8 + wave;       // 0 .. 393215
  const int mt = (int)(tile % 2048);
  const int nt = (int)(tile / 2048);
  const int m0 = mt * 16, n0 = nt * 16;

  v8f acc = {};
#pragma unroll
  for (int kk = 0; kk < Epad; kk += 32) {
    if (kk + 32 < Epad)
      __builtin_prefetch(Wih + (size_t)(n0 + (lane & 15)) * Epad + kk + 32, 0, 1);
    v16bf a = load_a_frag(Ae, Epad, m0, kk, lane);
    v16bf b = load_b_frag(Wih, Epad, n0, kk, lane);
    acc = __builtin_amdgcn_wmma_f32_16x16x32_bf16(false, a, false, b,
                                                  (short)0, acc, false, false);
  }
  store_c_bias(gx, G3H, m0, n0, lane, acc, b_ih[n0 + (lane & 15)]);
}

// ------------------------- device-wide barrier -------------------------------
__device__ __forceinline__ void grid_barrier(unsigned* bar, unsigned nwg) {
  __syncthreads();
  if (threadIdx.x == 0) {
    volatile unsigned* gen = bar + 1;
    unsigned g = *gen;
    __threadfence();
    if (atomicAdd(bar, 1u) == nwg - 1u) {
      bar[0] = 0u;
      __threadfence();
      atomicAdd((unsigned*)(bar + 1), 1u);
    } else {
      while (*gen == g) __builtin_amdgcn_s_sleep(1);
    }
  }
  __syncthreads();
}

// --------- phase 2: persistent GRU recurrence (512 sequential steps) ---------
// 96 WGs x 8 waves = 768 waves = (64/16) x (3072/16) tiles of gh.
__global__ void __launch_bounds__(256) k_gru(const float* __restrict__ gx,
                                             const __bf16* __restrict__ Whh,
                                             const float* __restrict__ b_hh,
                                             float* __restrict__ gh,
                                             float* __restrict__ hf,
                                             __bf16* __restrict__ hb,
                                             unsigned* bar) {
  const int lane = threadIdx.x & 31;
  const int wave = threadIdx.x >> 5;
  const int tile = blockIdx.x * 8 + wave;              // 0..767
  const int m0 = (tile & 3) * 16;
  const int n0 = (tile >> 2) * 16;
  const float bias = b_hh[n0 + (lane & 15)];
  const unsigned nwg = gridDim.x;

  for (int t = 0; t < Tlen; ++t) {
    // gh = h @ W_hh^T + b_hh : 32 wmma per wave, operands L2/L0-resident
    v8f acc = {};
#pragma unroll 8
    for (int kk = 0; kk < Hdim; kk += 32) {
      if (kk + 32 < Hdim)
        __builtin_prefetch(Whh + (size_t)(n0 + (lane & 15)) * Hdim + kk + 32, 0, 1);
      v16bf a = load_a_frag(hb, Hdim, m0, kk, lane);
      v16bf b = load_b_frag(Whh, Hdim, n0, kk, lane);
      acc = __builtin_amdgcn_wmma_f32_16x16x32_bf16(false, a, false, b,
                                                    (short)0, acc, false, false);
    }
    store_c_bias(gh, G3H, m0, n0, lane, acc, bias);

    grid_barrier(bar, nwg);

    // gates: h_new = (1-z)*n + z*h  over 64x1024 elements
    for (int idx = blockIdx.x * 256 + threadIdx.x; idx < Bsz * Hdim;
         idx += nwg * 256) {
      int bb = idx >> 10;                 // batch
      int j  = idx & (Hdim - 1);          // hidden index
      const float* gxr = gx + ((size_t)bb * Tlen + t) * G3H;
      const float* ghr = gh + (size_t)bb * G3H;
      float r = sigmoidf_(gxr[j]            + ghr[j]);
      float z = sigmoidf_(gxr[Hdim + j]     + ghr[Hdim + j]);
      float n = tanhf    (gxr[2 * Hdim + j] + r * ghr[2 * Hdim + j]);
      float hn = (1.0f - z) * n + z * hf[idx];
      hf[idx] = hn;
      hb[idx] = (__bf16)hn;
    }

    grid_barrier(bar, nwg);
  }
}

// ------------------------------ FC head --------------------------------------
__global__ void k_fc(const float* __restrict__ hf, const float* __restrict__ Wfc,
                     const float* __restrict__ b_fc, float* __restrict__ out) {
  int i = threadIdx.x;                                  // 0..319
  if (i >= Bsz * Odim) return;
  int b = i / Odim, o = i % Odim;
  float s = b_fc[o];
  const float* hr = hf + (size_t)b * Hdim;
  const float* wr = Wfc + (size_t)o * Hdim;
  for (int j = 0; j < Hdim; ++j) s += hr[j] * wr[j];
  out[i] = s;
}

// ------------------------------ launcher -------------------------------------
extern "C" void kernel_launch(void* const* d_in, const int* in_sizes, int n_in,
                              void* d_out, int out_size, void* d_ws, size_t ws_size,
                              hipStream_t stream) {
  const int*   x    = (const int*)  d_in[0];
  const float* emb  = (const float*)d_in[1];
  const float* Wih  = (const float*)d_in[2];
  const float* Whh  = (const float*)d_in[3];
  const float* b_ih = (const float*)d_in[4];
  const float* b_hh = (const float*)d_in[5];
  const float* Wfc  = (const float*)d_in[6];
  const float* b_fc = (const float*)d_in[7];
  float* out = (float*)d_out;

  char* w = (char*)d_ws;
  unsigned* bar   = (unsigned*)(w + OFF_BAR);
  float*    hf    = (float*)   (w + OFF_HF);
  __bf16*   hb    = (__bf16*)  (w + OFF_HB);
  float*    gh    = (float*)   (w + OFF_GH);
  __bf16*   WihP  = (__bf16*)  (w + OFF_WIH);
  __bf16*   WhhP  = (__bf16*)  (w + OFF_WHH);
  __bf16*   Ae    = (__bf16*)  (w + OFF_AEMB);
  float*    gx    = (float*)   (w + OFF_GX);

  k_init    <<<256, 256, 0, stream>>>(hf, hb, bar);
  k_pack_wih<<<1920, 256, 0, stream>>>(Wih, WihP);
  k_pack_whh<<<4096, 256, 0, stream>>>(Whh, WhhP);
  k_embed   <<<8192, 256, 0, stream>>>(x, emb, Ae);

  // gx GEMM: 2048 x 192 tiles, 8 waves/WG -> 49152 blocks
  k_gemm_gx <<<49152, 256, 0, stream>>>(Ae, WihP, b_ih, gx);

  // persistent recurrence: 96 co-resident WGs, 512 steps, 2 grid barriers/step
  k_gru     <<<NUM_WG_GRU, 256, 0, stream>>>(gx, WhhP, b_hh, gh, hf, hb, bar);

  k_fc      <<<1, 320, 0, stream>>>(hf, Wfc, b_fc, out);
}
